// SplineCNNUNet_V1_68410239091223
// MI455X (gfx1250) — matrix operand order repaired and checked by reading, not compiled, not run on hardware
//
#include <hip/hip_runtime.h>
#include <hip/hip_bf16.h>
#include <math.h>

// ---------------------------------------------------------------------------
// SplineCNN U-Net forward for MI455X (gfx1250), wave32 + WMMA bf16.
// Dense algebra: LDS-staged 64x64x64 block GEMM, v_wmma_f32_16x16x32_bf16.
// ---------------------------------------------------------------------------

typedef __attribute__((ext_vector_type(16))) __bf16 v16bf;
typedef __attribute__((ext_vector_type(8)))  float  v8f;
typedef __attribute__((ext_vector_type(8)))  unsigned v8u;

static __device__ __forceinline__ unsigned short f2bf_raw(float f) {
  unsigned u = __builtin_bit_cast(unsigned, f);
  unsigned r = u + 0x7FFFu + ((u >> 16) & 1u);   // round-to-nearest-even
  return (unsigned short)(r >> 16);
}
static __device__ __forceinline__ unsigned pack2bf(float a, float b) {
  return (unsigned)f2bf_raw(a) | ((unsigned)f2bf_raw(b) << 16);
}
static __device__ __forceinline__ int imin(int a, int b) { return a < b ? a : b; }

// ------------------------------ utility kernels ----------------------------

__global__ void k_zero(float* __restrict__ p, long long n) {
  long long i = (long long)blockIdx.x * blockDim.x + threadIdx.x;
  long long st = (long long)gridDim.x * blockDim.x;
  for (; i < n; i += st) p[i] = 0.0f;
}

// bw[e][8], wi[e][8] from positions; edges duplicated across batches.
__global__ void k_spline_prep(const float* __restrict__ pos, const int* __restrict__ e,
                              int Et, int nPer, float inv_patch,
                              float* __restrict__ bw, int* __restrict__ wi, int Etot) {
  int te = blockIdx.x * blockDim.x + threadIdx.x;
  if (te >= Etot) return;
  int b = te / Et, k = te - b * Et;
  int src = e[k] + b * nPer;
  int dst = e[Et + k] + b * nPer;
  float fr[3]; int k0[3];
#pragma unroll
  for (int d = 0; d < 3; ++d) {
    float ps = fminf(fmaxf((pos[dst * 3 + d] - pos[src * 3 + d]) * inv_patch + 0.5f, 0.0f), 1.0f);
    float v  = ps * 2.0f;                 // (K-1) = 2
    float fl = floorf(v);
    fl = fminf(fmaxf(fl, 0.0f), 1.0f);    // clip to K-2
    fr[d] = v - fl;
    k0[d] = (int)fl;
  }
  const int strides[3] = {1, 3, 9};
#pragma unroll
  for (int s = 0; s < 8; ++s) {
    float w = 1.0f; int widx = 0;
#pragma unroll
    for (int d = 0; d < 3; ++d) {
      int bit = (s >> d) & 1;
      w *= bit ? fr[d] : (1.0f - fr[d]);
      widx += (k0[d] + bit) * strides[d];
    }
    bw[te * 8 + s] = w;
    wi[te * 8 + s] = widx;
  }
}

__global__ void k_degree(const int* __restrict__ e, int Et, int nPer,
                         float* __restrict__ deg, int Etot) {
  int te = blockIdx.x * blockDim.x + threadIdx.x;
  if (te >= Etot) return;
  int b = te / Et, k = te - b * Et;
  int dst = e[Et + k] + b * nPer;
  atomicAdd(&deg[dst], 1.0f);
}

// acc[dst][wi][c] += bw * x[src][c]   (one block per edge)
__global__ void k_scatter(const float* __restrict__ xin, const int* __restrict__ e,
                          int Et, int nPer, const float* __restrict__ bw,
                          const int* __restrict__ wi, int cin, float* __restrict__ acc) {
  int te = blockIdx.x;
  int b = te / Et, k = te - b * Et;
  int src = e[k] + b * nPer;
  int dst = e[Et + k] + b * nPer;
  float w[8]; int ix[8];
#pragma unroll
  for (int s = 0; s < 8; ++s) { w[s] = bw[te * 8 + s]; ix[s] = wi[te * 8 + s]; }
  for (int c = threadIdx.x; c < cin; c += blockDim.x) {
    float xv = xin[(size_t)src * cin + c];
#pragma unroll
    for (int s = 0; s < 8; ++s)
      atomicAdd(&acc[((size_t)dst * 27 + ix[s]) * cin + c], w[s] * xv);
  }
}

// ------------------------------ WMMA GEMM ----------------------------------
// C(MxN,f32) = A(MxK,f32,rm) * Bm(KxN,f32,rm).  bf16 operands, f32 acc.
// Block: 128 threads (4 waves). Tile: 64x64 output, K staged in 64-slabs
// through LDS (A row-major bf16, B transposed bf16).  Wave w computes rows
// [16w,16w+16) x all 64 cols => 8 WMMAs per K-slab per wave.
__global__ __launch_bounds__(128) void k_gemm_wmma(
    const float* __restrict__ A, const float* __restrict__ Bm,
    float* __restrict__ C, int M, int N, int Kd) {
  __shared__ __align__(16) unsigned shA[64 * 32];  // [row][kdword] 2 bf16/dword
  __shared__ __align__(16) unsigned shB[64 * 32];  // [col][kdword] (transposed)

  const int tid  = threadIdx.x;
  const int lane = tid & 31;
  const int wave = tid >> 5;
  const int tilesN = (N + 63) >> 6;
  const int bm = blockIdx.x / tilesN;
  const int bn = blockIdx.x - bm * tilesN;
  const int row0 = bm * 64, col0 = bn * 64;
  const int r16 = lane & 15, hi = lane >> 4;

  v8f acc0 = {}, acc1 = {}, acc2 = {}, acc3 = {};

  for (int k0 = 0; k0 < Kd; k0 += 64) {
    // ---- stage A slab: 64 rows x 64 k (f32 -> packed bf16), branch-free ----
    for (int l = tid; l < 64 * 32; l += 128) {
      int r = l >> 5, d = l & 31;
      int k = k0 + 2 * d;
      int gr = row0 + r;
      size_t base = (size_t)imin(gr, M - 1) * Kd;
      float a0 = A[base + imin(k,     Kd - 1)];
      float a1 = A[base + imin(k + 1, Kd - 1)];
      a0 = (gr < M && k     < Kd) ? a0 : 0.0f;
      a1 = (gr < M && k + 1 < Kd) ? a1 : 0.0f;
      shA[l] = pack2bf(a0, a1);
    }
    // ---- stage B slab transposed: 64 cols x 64 k ----
    for (int l = tid; l < 64 * 32; l += 128) {
      int cc = l >> 5, d = l & 31;
      int k = k0 + 2 * d;
      int gc = col0 + cc;
      int gcc = imin(gc, N - 1);
      float b0 = Bm[(size_t)imin(k,     Kd - 1) * N + gcc];
      float b1 = Bm[(size_t)imin(k + 1, Kd - 1) * N + gcc];
      b0 = (gc < N && k     < Kd) ? b0 : 0.0f;
      b1 = (gc < N && k + 1 < Kd) ? b1 : 0.0f;
      shB[l] = pack2bf(b0, b1);
    }
    __syncthreads();

#pragma unroll
    for (int kh = 0; kh < 2; ++kh) {
      // A fragment (16x32 bf16): lane r16 = row, hi selects K half-groups.
      // dwords: row*32 + kh*16 + hi*4 + {0..3}  and  +8 + {0..3}
      const int abase = (wave * 16 + r16) * 32 + kh * 16 + hi * 4;
      uint4 a0 = *(const uint4*)&shA[abase];
      uint4 a1 = *(const uint4*)&shA[abase + 8];
      v8u ua; ua[0]=a0.x; ua[1]=a0.y; ua[2]=a0.z; ua[3]=a0.w;
              ua[4]=a1.x; ua[5]=a1.y; ua[6]=a1.z; ua[7]=a1.w;
      v16bf afr = __builtin_bit_cast(v16bf, ua);
#pragma unroll
      for (int t = 0; t < 4; ++t) {
        // B fragment (32x16 bf16): lane r16 = col, hi selects K half-groups.
        // dwords: col*32 + kh*16 + hi*4 + {0..3}  and  +8 + {0..3}
        const int bbase = (t * 16 + r16) * 32 + kh * 16 + hi * 4;
        uint4 b0 = *(const uint4*)&shB[bbase];
        uint4 b1 = *(const uint4*)&shB[bbase + 8];
        v8u ub; ub[0]=b0.x; ub[1]=b0.y; ub[2]=b0.z; ub[3]=b0.w;
                ub[4]=b1.x; ub[5]=b1.y; ub[6]=b1.z; ub[7]=b1.w;
        v16bf bfr = __builtin_bit_cast(v16bf, ub);
        v8f* pa = (t == 0) ? &acc0 : (t == 1) ? &acc1 : (t == 2) ? &acc2 : &acc3;
        *pa = __builtin_amdgcn_wmma_f32_16x16x32_bf16(false, afr, false, bfr,
                                                      (short)0, *pa, false, false);
      }
    }
    __syncthreads();
  }

  // ---- store: C/D layout VGPR v -> row v + 8*hi ----
#pragma unroll
  for (int t = 0; t < 4; ++t) {
    v8f a = (t == 0) ? acc0 : (t == 1) ? acc1 : (t == 2) ? acc2 : acc3;
    int ccol = col0 + t * 16 + r16;
    if (ccol < N) {
#pragma unroll
      for (int v = 0; v < 8; ++v) {
        int r = row0 + wave * 16 + hi * 8 + v;
        if (r < M) C[(size_t)r * N + ccol] = a[v];
      }
    }
  }
}

// ------------------------------ epilogues ----------------------------------
// act: 0=none, 1=elu, 2=relu
__global__ void k_conv_epilogue(const float* __restrict__ g1, const float* __restrict__ g2,
                                const float* __restrict__ bias, const float* __restrict__ deg,
                                float* __restrict__ out, int n, int cout, int act) {
  int t = blockIdx.x * blockDim.x + threadIdx.x;
  if (t >= n * cout) return;
  int row = t / cout, ch = t - row * cout;
  float d = fmaxf(deg[row], 1.0f);
  float v = g1[t] / d + g2[t] + bias[ch];
  if (act == 1) v = (v > 0.0f) ? v : expm1f(v);
  else if (act == 2) v = fmaxf(v, 0.0f);
  out[t] = v;
}

__global__ void k_bias_act(const float* __restrict__ g, const float* __restrict__ bias,
                           float* __restrict__ out, int n, int c, int act) {
  int t = blockIdx.x * blockDim.x + threadIdx.x;
  if (t >= n * c) return;
  int ch = t % c;
  float v = g[t] + bias[ch];
  if (act == 1) v = (v > 0.0f) ? v : expm1f(v);
  else if (act == 2) v = fmaxf(v, 0.0f);
  out[t] = v;
}

// batchnorm over rows, one block per channel; act 2 = relu
__global__ void k_bnorm(const float* __restrict__ z, const float* __restrict__ g,
                        const float* __restrict__ be, float* __restrict__ out,
                        int n, int c, int act) {
  int ch = blockIdx.x;
  __shared__ float s1[256], s2[256];
  float a = 0.0f, b = 0.0f;
  for (int i = threadIdx.x; i < n; i += blockDim.x) {
    float v = z[(size_t)i * c + ch];
    a += v; b += v * v;
  }
  s1[threadIdx.x] = a; s2[threadIdx.x] = b;
  __syncthreads();
  for (int off = 128; off > 0; off >>= 1) {
    if ((int)threadIdx.x < off) { s1[threadIdx.x] += s1[threadIdx.x + off]; s2[threadIdx.x] += s2[threadIdx.x + off]; }
    __syncthreads();
  }
  float m   = s1[0] / (float)n;
  float var = s2[0] / (float)n - m * m;
  float sc  = rsqrtf(var + 1e-5f) * g[ch];
  float bb  = be[ch];
  for (int i = threadIdx.x; i < n; i += blockDim.x) {
    float y = (z[(size_t)i * c + ch] - m) * sc + bb;
    if (act == 2) y = fmaxf(y, 0.0f);
    out[(size_t)i * c + ch] = y;
  }
}

// pooled[b][ch] = max over nPer rows of segment b (2 segments)
__global__ void k_segmax(const float* __restrict__ g, float* __restrict__ pooled,
                         int nPer, int c) {
  int t = blockIdx.x * blockDim.x + threadIdx.x;
  if (t >= 2 * c) return;
  int b = t / c, ch = t - b * c;
  float m = -3.0e38f;
  for (int i = 0; i < nPer; ++i) m = fmaxf(m, g[(size_t)(b * nPer + i) * c + ch]);
  pooled[t] = m;
}

// dst[j][:] = src[idx[j % outPer] + (j/outPer)*nPer][:]
__global__ void k_gather_rows(const float* __restrict__ src, const int* __restrict__ idx,
                              int nPer, int outPer, int c, float* __restrict__ dst, int total) {
  int t = blockIdx.x * blockDim.x + threadIdx.x;
  if (t >= total) return;
  int j = t / c, ch = t - j * c;
  int b = j / outPer, r = idx[j - b * outPer] + b * nPer;
  dst[t] = src[(size_t)r * c + ch];
}

__global__ void k_concat2(const float* __restrict__ a, int ca,
                          const float* __restrict__ b, int cb,
                          float* __restrict__ dst, int n) {
  int c = ca + cb;
  int t = blockIdx.x * blockDim.x + threadIdx.x;
  if (t >= n * c) return;
  int row = t / c, ch = t - row * c;
  dst[t] = (ch < ca) ? a[(size_t)row * ca + ch] : b[(size_t)row * cb + (ch - ca)];
}

// dst[j] = [pooled[j/nPer][:c1], hb[j][:c2]]
__global__ void k_bcast_concat(const float* __restrict__ pooled, const float* __restrict__ hb,
                               float* __restrict__ dst, int nPer, int c1, int c2, int n) {
  int c = c1 + c2;
  int t = blockIdx.x * blockDim.x + threadIdx.x;
  if (t >= n * c) return;
  int row = t / c, ch = t - row * c;
  dst[t] = (ch < c1) ? pooled[(size_t)(row / nPer) * c1 + ch]
                     : hb[(size_t)row * c2 + (ch - c1)];
}

// kNN(k=3) inverse-distance interpolation; one block per destination point.
__global__ void k_knn3(const float* __restrict__ xs, const float* __restrict__ ps,
                       const float* __restrict__ pd, float* __restrict__ y,
                       int ns, int nd, int c) {
  int j = blockIdx.x;                 // 0 .. B*nd-1
  int b = j / nd;
  float px = pd[j * 3], py = pd[j * 3 + 1], pz = pd[j * 3 + 2];
  float bd[3] = {1e30f, 1e30f, 1e30f}; int bi[3] = {0, 0, 0};
  for (int i = threadIdx.x; i < ns; i += blockDim.x) {
    int row = b * ns + i;
    float dx = px - ps[row * 3], dy = py - ps[row * 3 + 1], dz = pz - ps[row * 3 + 2];
    float d2 = dx * dx + dy * dy + dz * dz;
    if (d2 < bd[2]) {
      if (d2 < bd[1]) {
        bd[2] = bd[1]; bi[2] = bi[1];
        if (d2 < bd[0]) { bd[1] = bd[0]; bi[1] = bi[0]; bd[0] = d2; bi[0] = row; }
        else            { bd[1] = d2;  bi[1] = row; }
      } else { bd[2] = d2; bi[2] = row; }
    }
  }
  __shared__ float sd[384]; __shared__ int si[384];
  __shared__ float fw[3];  __shared__ int fi[3]; __shared__ float fwsum;
#pragma unroll
  for (int s = 0; s < 3; ++s) { sd[threadIdx.x * 3 + s] = bd[s]; si[threadIdx.x * 3 + s] = bi[s]; }
  __syncthreads();
  if (threadIdx.x == 0) {
    float gd[3] = {1e30f, 1e30f, 1e30f}; int gi[3] = {0, 0, 0};
    int tot = blockDim.x * 3;
    for (int t = 0; t < tot; ++t) {
      float d2 = sd[t]; int row = si[t];
      if (d2 < gd[2]) {
        if (d2 < gd[1]) {
          gd[2] = gd[1]; gi[2] = gi[1];
          if (d2 < gd[0]) { gd[1] = gd[0]; gi[1] = gi[0]; gd[0] = d2; gi[0] = row; }
          else            { gd[1] = d2;  gi[1] = row; }
        } else { gd[2] = d2; gi[2] = row; }
      }
    }
    float ws = 0.0f;
#pragma unroll
    for (int s = 0; s < 3; ++s) { float w = 1.0f / fmaxf(gd[s], 1e-16f); fw[s] = w; fi[s] = gi[s]; ws += w; }
    fwsum = ws;
  }
  __syncthreads();
  float w0 = fw[0], w1 = fw[1], w2 = fw[2], inv = 1.0f / fwsum;
  int i0 = fi[0], i1 = fi[1], i2 = fi[2];
  for (int ch = threadIdx.x; ch < c; ch += blockDim.x) {
    float v = w0 * xs[(size_t)i0 * c + ch] + w1 * xs[(size_t)i1 * c + ch] + w2 * xs[(size_t)i2 * c + ch];
    y[(size_t)j * c + ch] = v * inv;
  }
}

// ------------------------------ host helpers -------------------------------

static inline void zero_buf(float* p, long long n, hipStream_t s) {
  long long blocks = (n + 255) / 256; if (blocks > 65536) blocks = 65536;
  k_zero<<<(unsigned)blocks, 256, 0, s>>>(p, n);
}

static inline void gemm(const float* A, const float* Bm, float* C,
                        int M, int N, int K, hipStream_t s) {
  int tiles = ((M + 63) / 64) * ((N + 63) / 64);
  k_gemm_wmma<<<tiles, 128, 0, s>>>(A, Bm, C, M, N, K);
}

struct CP { const float *W, *Wr, *b; };

static inline void spline_conv(const float* xin, int cin, int cout,
                               const int* e, int Et, int nPer, int n,
                               const float* bw, const int* wi, const float* deg,
                               CP p, float* acc, float* g1, float* g2,
                               float* out, int act, hipStream_t s) {
  long long accN = (long long)n * 27 * cin;
  zero_buf(acc, accN, s);
  int Etot = 2 * Et;
  int th = (cin <= 32) ? 32 : ((cin <= 64) ? 64 : 128);
  k_scatter<<<Etot, th, 0, s>>>(xin, e, Et, nPer, bw, wi, cin, acc);
  gemm(acc, p.W, g1, n, cout, 27 * cin, s);
  gemm(xin, p.Wr, g2, n, cout, cin, s);
  int tot = n * cout;
  k_conv_epilogue<<<(tot + 255) / 256, 256, 0, s>>>(g1, g2, p.b, deg, out, n, cout, act);
}

// ------------------------------ entry point --------------------------------

extern "C" void kernel_launch(void* const* d_in, const int* in_sizes, int n_in,
                              void* d_out, int out_size, void* d_ws, size_t ws_size,
                              hipStream_t stream) {
  (void)in_sizes; (void)n_in; (void)out_size; (void)ws_size;
  const int N0 = 4096, N1 = 1024, N2 = 256;
  const int E0 = 49152, E1 = 12288, E2 = 3072;
  const int n0 = 2 * N0, n1 = 2 * N1, n2 = 2 * N2;

  const float* x   = (const float*)d_in[0];
  const float* pos = (const float*)d_in[1];
  int pi = 2;
  struct LP { const float *W, *b; };
  struct BP { const float *g, *be; };
  auto getc = [&]() { CP c{(const float*)d_in[pi], (const float*)d_in[pi + 1], (const float*)d_in[pi + 2]}; pi += 3; return c; };
  auto getl = [&]() { LP l{(const float*)d_in[pi], (const float*)d_in[pi + 1]}; pi += 2; return l; };
  auto getb = [&]() { BP b{(const float*)d_in[pi], (const float*)d_in[pi + 1]}; pi += 2; return b; };
  CP sa1[3] = {getc(), getc(), getc()};
  CP sa2[3] = {getc(), getc(), getc()};
  CP bnc[3] = {getc(), getc(), getc()};
  LP gl[3]  = {getl(), getl(), getl()};
  BP gb[2]  = {getb(), getb()};
  CP fpa[3] = {getc(), getc(), getc()};   // fp2
  CP fpb[3] = {getc(), getc(), getc()};   // fp1
  LP hd[2]  = {getl(), getl()};
  // d_in[pi] = batch (unused: batch == row / Nlevel by construction)
  const int* idx0 = (const int*)d_in[pi + 1];
  const int* idx1 = (const int*)d_in[pi + 2];
  const int* e0   = (const int*)d_in[pi + 3];
  const int* e1   = (const int*)d_in[pi + 4];
  const int* e2   = (const int*)d_in[pi + 5];

  // ---- workspace bump allocator (f32 elements, 256B aligned chunks) ----
  char* wsp = (char*)d_ws;
  auto af = [&](size_t elems) -> float* { float* p = (float*)wsp; wsp += ((elems * 4 + 255) / 256) * 256; return p; };
  auto ai = [&](size_t elems) -> int*   { int*   p = (int*)wsp;   wsp += ((elems * 4 + 255) / 256) * 256; return p; };

  float* bw0 = af((size_t)2 * E0 * 8); int* wi0 = ai((size_t)2 * E0 * 8);
  float* bw1 = af((size_t)2 * E1 * 8); int* wi1 = ai((size_t)2 * E1 * 8);
  float* bw2 = af((size_t)2 * E2 * 8); int* wi2 = ai((size_t)2 * E2 * 8);
  float* deg0 = af(n0); float* deg1 = af(n1); float* deg2 = af(n2);
  float* hA   = af((size_t)n0 * 32);  float* hB  = af((size_t)n0 * 32);
  float* x1   = af((size_t)n1 * 32);  float* pos1 = af((size_t)n1 * 3);
  float* h1A  = af((size_t)n1 * 128); float* h1B = af((size_t)n1 * 128);
  float* x2   = af((size_t)n2 * 128); float* pos2 = af((size_t)n2 * 3);
  float* hbA  = af((size_t)n2 * 128); float* hbB = af((size_t)n2 * 128);
  float* gcat = af((size_t)n2 * 131);
  float* gA   = af((size_t)n2 * 1024); float* gB = af((size_t)n2 * 1024);
  float* pooled = af((size_t)2 * 1024);
  float* xg   = af((size_t)n2 * 1152);
  float* knnA = af((size_t)n1 * 1152);
  float* xk   = af((size_t)n1 * 1184);
  float* f2A  = af((size_t)n1 * 256); float* f2B = af((size_t)n1 * 256);
  float* knn0 = af((size_t)n0 * 128);
  float* xk0  = af((size_t)n0 * 134);
  float* f1A  = af((size_t)n0 * 128); float* f1B = af((size_t)n0 * 128);
  float* hh   = af((size_t)n0 * 64);
  float* g1   = af((size_t)n0 * 128);   // GEMM scratch (max M*N)
  float* g2   = af((size_t)n0 * 128);
  float* acc  = af((size_t)n1 * 27 * 1184);  // largest spline-conv accumulator

  // ---- spline basis + degrees per level ----
  k_spline_prep<<<(2 * E0 + 255) / 256, 256, 0, stream>>>(pos,  e0, E0, N0, 2.0f, bw0, wi0, 2 * E0);
  zero_buf(deg0, n0, stream);
  k_degree<<<(2 * E0 + 255) / 256, 256, 0, stream>>>(e0, E0, N0, deg0, 2 * E0);

  // ---- SA1: 6 -> 16 -> 16 -> 32 (level 0) ----
  spline_conv(x,  6, 16, e0, E0, N0, n0, bw0, wi0, deg0, sa1[0], acc, g1, g2, hA, 1, stream);
  spline_conv(hA, 16, 16, e0, E0, N0, n0, bw0, wi0, deg0, sa1[1], acc, g1, g2, hB, 1, stream);
  spline_conv(hB, 16, 32, e0, E0, N0, n0, bw0, wi0, deg0, sa1[2], acc, g1, g2, hA, 1, stream);

  // ---- pool to level 1 ----
  k_gather_rows<<<(n1 * 32 + 255) / 256, 256, 0, stream>>>(hA,  idx0, N0, N1, 32, x1,   n1 * 32);
  k_gather_rows<<<(n1 * 3  + 255) / 256, 256, 0, stream>>>(pos, idx0, N0, N1, 3,  pos1, n1 * 3);
  k_spline_prep<<<(2 * E1 + 255) / 256, 256, 0, stream>>>(pos1, e1, E1, N1, 1.0f, bw1, wi1, 2 * E1);
  zero_buf(deg1, n1, stream);
  k_degree<<<(2 * E1 + 255) / 256, 256, 0, stream>>>(e1, E1, N1, deg1, 2 * E1);

  // ---- SA2: 32 -> 64 -> 64 -> 128 (level 1) ----
  spline_conv(x1,  32,  64, e1, E1, N1, n1, bw1, wi1, deg1, sa2[0], acc, g1, g2, h1A, 1, stream);
  spline_conv(h1A, 64,  64, e1, E1, N1, n1, bw1, wi1, deg1, sa2[1], acc, g1, g2, h1B, 1, stream);
  spline_conv(h1B, 64, 128, e1, E1, N1, n1, bw1, wi1, deg1, sa2[2], acc, g1, g2, h1A, 1, stream);

  // ---- pool to level 2 ----
  k_gather_rows<<<(n2 * 128 + 255) / 256, 256, 0, stream>>>(h1A,  idx1, N1, N2, 128, x2,   n2 * 128);
  k_gather_rows<<<(n2 * 3   + 255) / 256, 256, 0, stream>>>(pos1, idx1, N1, N2, 3,   pos2, n2 * 3);
  k_spline_prep<<<(2 * E2 + 255) / 256, 256, 0, stream>>>(pos2, e2, E2, N2, 0.5f, bw2, wi2, 2 * E2);
  zero_buf(deg2, n2, stream);
  k_degree<<<(2 * E2 + 255) / 256, 256, 0, stream>>>(e2, E2, N2, deg2, 2 * E2);

  // ---- bottleneck convs: 128 -> 128 -> 128 -> 128 (level 2) ----
  spline_conv(x2,  128, 128, e2, E2, N2, n2, bw2, wi2, deg2, bnc[0], acc, g1, g2, hbA, 1, stream);
  spline_conv(hbA, 128, 128, e2, E2, N2, n2, bw2, wi2, deg2, bnc[1], acc, g1, g2, hbB, 1, stream);
  spline_conv(hbB, 128, 128, e2, E2, N2, n2, bw2, wi2, deg2, bnc[2], acc, g1, g2, hbA, 1, stream);

  // ---- global MLP: concat(x2,pos2) -> 256 -> 512 -> 1024, BN+ReLU between ----
  k_concat2<<<(n2 * 131 + 255) / 256, 256, 0, stream>>>(x2, 128, pos2, 3, gcat, n2);
  gemm(gcat, gl[0].W, g1, n2, 256, 131, stream);
  k_bias_act<<<(n2 * 256 + 255) / 256, 256, 0, stream>>>(g1, gl[0].b, gA, n2, 256, 0);
  k_bnorm<<<256, 256, 0, stream>>>(gA, gb[0].g, gb[0].be, gB, n2, 256, 2);
  gemm(gB, gl[1].W, g1, n2, 512, 256, stream);
  k_bias_act<<<(n2 * 512 + 255) / 256, 256, 0, stream>>>(g1, gl[1].b, gA, n2, 512, 0);
  k_bnorm<<<512, 256, 0, stream>>>(gA, gb[1].g, gb[1].be, gB, n2, 512, 2);
  gemm(gB, gl[2].W, g1, n2, 1024, 512, stream);
  k_bias_act<<<(n2 * 1024 + 255) / 256, 256, 0, stream>>>(g1, gl[2].b, gA, n2, 1024, 0);

  // ---- global max pool + broadcast concat with bottleneck features ----
  k_segmax<<<(2 * 1024 + 255) / 256, 256, 0, stream>>>(gA, pooled, N2, 1024);
  k_bcast_concat<<<(n2 * 1152 + 255) / 256, 256, 0, stream>>>(pooled, hbA, xg, N2, 1024, 128, n2);

  // ---- FP2: knn-interp level2 -> level1, concat skip, convs 1184->256->256->128 ----
  k_knn3<<<n1, 128, 0, stream>>>(xg, pos2, pos1, knnA, N2, N1, 1152);
  k_concat2<<<(n1 * 1184 + 255) / 256, 256, 0, stream>>>(knnA, 1152, x1, 32, xk, n1);
  spline_conv(xk,  1184, 256, e1, E1, N1, n1, bw1, wi1, deg1, fpa[0], acc, g1, g2, f2A, 1, stream);
  spline_conv(f2A,  256, 256, e1, E1, N1, n1, bw1, wi1, deg1, fpa[1], acc, g1, g2, f2B, 1, stream);
  spline_conv(f2B,  256, 128, e1, E1, N1, n1, bw1, wi1, deg1, fpa[2], acc, g1, g2, f2A, 1, stream);

  // ---- FP1: knn-interp level1 -> level0, concat skip, convs 134->128->128->128 ----
  k_knn3<<<n0, 128, 0, stream>>>(f2A, pos1, pos, knn0, N1, N0, 128);
  k_concat2<<<(n0 * 134 + 255) / 256, 256, 0, stream>>>(knn0, 128, x, 6, xk0, n0);
  spline_conv(xk0, 134, 128, e0, E0, N0, n0, bw0, wi0, deg0, fpb[0], acc, g1, g2, f1A, 1, stream);
  spline_conv(f1A, 128, 128, e0, E0, N0, n0, bw0, wi0, deg0, fpb[1], acc, g1, g2, f1B, 1, stream);
  spline_conv(f1B, 128, 128, e0, E0, N0, n0, bw0, wi0, deg0, fpb[2], acc, g1, g2, f1A, 1, stream);

  // ---- head: 128 -> 64 (relu) -> 3 ----
  gemm(f1A, hd[0].W, g1, n0, 64, 128, stream);
  k_bias_act<<<(n0 * 64 + 255) / 256, 256, 0, stream>>>(g1, hd[0].b, hh, n0, 64, 2);
  gemm(hh, hd[1].W, g1, n0, 3, 64, stream);
  k_bias_act<<<(n0 * 3 + 255) / 256, 256, 0, stream>>>(g1, hd[1].b, (float*)d_out, n0, 3, 0);
}
